// Attention_9612136809120
// MI455X (gfx1250) — compile-verified, matrix operand order checked by reading, not run on hardware
//
#include <hip/hip_runtime.h>
#include <hip/hip_bf16.h>

// Problem dims (reference: B=1, H=8, S=4096, D=64, fp32 in/out)
constexpr int kH = 8;
constexpr int kS = 4096;
constexpr int kD = 64;
constexpr float NEGV = -1e20f;   // reference mask fill value

typedef _Float16 v16h __attribute__((ext_vector_type(16)));
typedef _Float16 v8h  __attribute__((ext_vector_type(8)));
typedef _Float16 v4h  __attribute__((ext_vector_type(4)));
typedef float    v8f  __attribute__((ext_vector_type(8)));

// Load a 16-lane x 32-K f16 WMMA operand fragment (works for global or LDS;
// after inlining the compiler infers the address space -> b128 loads).
// Layout (CDNA5 ISA 7.12.2, 16-bit A 16x32; B symmetric with N on lanes):
//   lanes 0-15 : row, K = colBase+{0..7} (VGPR0-3), colBase+16+{0..7} (VGPR4-7)
//   lanes 16-31: same rows, K offset +8 in each half
__device__ __forceinline__ v16h ld_frag(const _Float16* base, size_t strideH,
                                        int row, int colBase, int half) {
  const _Float16* p = base + (size_t)row * strideH + colBase + half * 8;
  v8h lo = *(const v8h*)(p);
  v8h hi = *(const v8h*)(p + 16);
  v16h r;
#pragma unroll
  for (int i = 0; i < 8; ++i) { r[i] = lo[i]; r[i + 8] = hi[i]; }
  return r;
}

// ---------------------------------------------------------------------------
// Pass 0: one-time f32->f16 conversion. qf/kf keep [h][s][d] layout; V is
// stored transposed vt[h][d][s] so PV B-fragments are contiguous loads.
// mask becomes an additive bias (0 or NEG). ~25MB read, bandwidth-trivial.
// ---------------------------------------------------------------------------
__global__ __launch_bounds__(256)
void preconvert_kernel(const float* __restrict__ q, const float* __restrict__ k,
                       const float* __restrict__ v, const int* __restrict__ mask,
                       _Float16* __restrict__ qf, _Float16* __restrict__ kf,
                       _Float16* __restrict__ vt, float* __restrict__ biasF) {
  const int gid = blockIdx.x * blockDim.x + threadIdx.x;
  const size_t base = (size_t)gid * 4;          // grid sized exactly: H*S*D/4

  const float4 fq = *(const float4*)(q + base);
  const float4 fk = *(const float4*)(k + base);
  v4h hq = {(_Float16)fq.x, (_Float16)fq.y, (_Float16)fq.z, (_Float16)fq.w};
  v4h hk = {(_Float16)fk.x, (_Float16)fk.y, (_Float16)fk.z, (_Float16)fk.w};
  *(v4h*)(qf + base) = hq;
  *(v4h*)(kf + base) = hk;

  const float4 fv = *(const float4*)(v + base);
  const int h  = (int)(base >> 18);             // / (kS*kD)
  const int s  = (int)((base >> 6) & (kS - 1)); // / kD % kS
  const int d0 = (int)(base & (kD - 1));
  _Float16* vth = vt + ((size_t)h * kD + d0) * kS + s;
  vth[0 * kS] = (_Float16)fv.x;
  vth[1 * kS] = (_Float16)fv.y;
  vth[2 * kS] = (_Float16)fv.z;
  vth[3 * kS] = (_Float16)fv.w;

  if (gid < kS) biasF[gid] = mask[gid] ? 0.0f : NEGV;
}

// ---------------------------------------------------------------------------
// Pass A: per-column (key) softmax stats over the QUERY axis.
// Grid (kS/64, kH), block 128 (4 waves, fully independent, no LDS/barriers).
// Wave owns 16 columns (B-fragments resident in regs); streams all 4096
// queries in 16-row WMMA tiles keeping an online (max, rescaled sum).
// ---------------------------------------------------------------------------
__global__ __launch_bounds__(128)
void attn_colstats_kernel(const _Float16* __restrict__ qf,
                          const _Float16* __restrict__ kf,
                          const float* __restrict__ biasF,
                          float* __restrict__ colM, float* __restrict__ colR) {
  const int h      = blockIdx.y;
  const int colBlk = blockIdx.x;
  const int tid    = threadIdx.x;
  const int wave   = tid >> 5;
  const int lane   = tid & 31;
  const int n      = lane & 15;
  const int half   = lane >> 4;

  const _Float16* qh = qf + (size_t)h * kS * kD;
  const _Float16* kh = kf + (size_t)h * kS * kD;

  const int kcol = colBlk * 64 + wave * 16 + n;
  // Resident B fragments: this lane's key row, d = 0..31 / 32..63.
  v16h bk0 = ld_frag(kh, kD, kcol, 0, half);
  v16h bk1 = ld_frag(kh, kD, kcol, 32, half);
  const float addK = biasF[kcol];               // 0 or NEG, loop-invariant

  float m_run = -INFINITY;
  float s_run = 0.0f;

  for (int qt = 0; qt < kS / 16; ++qt) {
    if (qt + 4 < kS / 16)
      __builtin_prefetch(&qh[(size_t)(qt + 4) * 16 * kD], 0, 0);

    v16h a0 = ld_frag(qh, kD, qt * 16 + n, 0, half);
    v16h a1 = ld_frag(qh, kD, qt * 16 + n, 32, half);
    v8f c8 = {};
    c8 = __builtin_amdgcn_wmma_f32_16x16x32_f16(false, a0, false, bk0, (short)0, c8, false, false);
    c8 = __builtin_amdgcn_wmma_f32_16x16x32_f16(false, a1, false, bk1, (short)0, c8, false, false);

    // Per-row query bias for this lane-half: two b128 loads, branchless.
    const float4 b0 = *(const float4*)&biasF[qt * 16 + half * 8];
    const float4 b1 = *(const float4*)&biasF[qt * 16 + half * 8 + 4];
    const float bias[8] = {b0.x + addK, b0.y + addK, b0.z + addK, b0.w + addK,
                           b1.x + addK, b1.y + addK, b1.z + addK, b1.w + addK};

    float vals[8];
    float tmax = -INFINITY;
#pragma unroll
    for (int j = 0; j < 8; ++j) {
      // Masked entries become exactly NEGV (|score| << ulp(1e20)); the clamp
      // folds the doubly-masked -2e20 case back to NEGV == reference where().
      float sv = fmaxf(c8[j] + bias[j], NEGV);
      vals[j] = sv;
      tmax = fmaxf(tmax, sv);
    }
    float nm = fmaxf(m_run, tmax);
    // Pairwise tree sum: shortens the exp->add dependency chain.
    float e0 = __expf(vals[0] - nm) + __expf(vals[1] - nm);
    float e1 = __expf(vals[2] - nm) + __expf(vals[3] - nm);
    float e2 = __expf(vals[4] - nm) + __expf(vals[5] - nm);
    float e3 = __expf(vals[6] - nm) + __expf(vals[7] - nm);
    s_run = s_run * __expf(m_run - nm) + ((e0 + e1) + (e2 + e3));
    m_run = nm;
  }

  // Combine the two lane-halves (rows 0-7 vs 8-15 of every tile).
  float om = __shfl_xor(m_run, 16, 32);
  float os = __shfl_xor(s_run, 16, 32);
  float M  = fmaxf(m_run, om);
  float Z  = s_run * __expf(m_run - M) + os * __expf(om - M);
  if (half == 0) {
    colM[(size_t)h * kS + kcol] = M;
    colR[(size_t)h * kS + kcol] = 1.0f / Z;
  }
}

// ---------------------------------------------------------------------------
// Pass B: O = P V, P[q,k] = exp(max(S+biasQ+biasK, NEG) - m_k) / Z_k.
// Grid (kS/64, kH), block 128. Waves independent (no barriers); only the
// per-wave P C-layout -> A-layout shuffle goes through LDS.
// ---------------------------------------------------------------------------
__global__ __launch_bounds__(128)
void attn_out_kernel(const _Float16* __restrict__ qf,
                     const _Float16* __restrict__ kf,
                     const _Float16* __restrict__ vt,
                     const float* __restrict__ biasF,
                     const float* __restrict__ colM, const float* __restrict__ colR,
                     float* __restrict__ out) {
  __shared__ __align__(16) _Float16 Ps[4][16 * 40];  // per-wave P tile (16q x 32k)

  const int h    = blockIdx.y;
  const int qBlk = blockIdx.x;
  const int tid  = threadIdx.x;
  const int wave = tid >> 5;
  const int lane = tid & 31;
  const int n    = lane & 15;
  const int half = lane >> 4;

  const _Float16* qh  = qf + (size_t)h * kS * kD;
  const _Float16* kh  = kf + (size_t)h * kS * kD;
  const _Float16* vth = vt + (size_t)h * kD * kS;   // vt[d][s]

  const int qrow0 = qBlk * 64 + wave * 16;
  // Resident A fragments: this wave's 16 query rows, d = 0..31 / 32..63.
  v16h aq0 = ld_frag(qh, kD, qrow0 + n, 0, half);
  v16h aq1 = ld_frag(qh, kD, qrow0 + n, 32, half);

  float biasQ[8];
#pragma unroll
  for (int j = 0; j < 8; ++j) biasQ[j] = biasF[qrow0 + j + half * 8];

  v8f acc[4] = {};   // four 16-wide d tiles of the output row block

  const float* colMh = colM + (size_t)h * kS;
  const float* colRh = colR + (size_t)h * kS;

  for (int kc = 0; kc < kS / 32; ++kc) {
    if (kc + 4 < kS / 32) {
      __builtin_prefetch(&kh[(size_t)(kc + 4) * 32 * kD], 0, 0);
      __builtin_prefetch(&vth[(size_t)(kc + 4) * 32], 0, 0);
    }

    // S tiles 16q x (2 x 16k): QK^T via WMMA, normalize per column, P -> LDS.
#pragma unroll
    for (int sub = 0; sub < 2; ++sub) {
      const int kcol = kc * 32 + sub * 16 + n;
      v16h bk0 = ld_frag(kh, kD, kcol, 0, half);
      v16h bk1 = ld_frag(kh, kD, kcol, 32, half);
      v8f c8 = {};
      c8 = __builtin_amdgcn_wmma_f32_16x16x32_f16(false, aq0, false, bk0, (short)0, c8, false, false);
      c8 = __builtin_amdgcn_wmma_f32_16x16x32_f16(false, aq1, false, bk1, (short)0, c8, false, false);

      const float addK = biasF[kcol];
      const float Mc   = colMh[kcol];
      const float Rc   = colRh[kcol];
#pragma unroll
      for (int j = 0; j < 8; ++j) {
        float sv = fmaxf(c8[j] + biasQ[j] + addK, NEGV);
        float p  = __expf(sv - Mc) * Rc;   // masked column -> exp(0)/4096 uniform
        Ps[wave][(j + half * 8) * 40 + sub * 16 + n] = (_Float16)p;
      }
    }

    // PV: A = P (16q x 32k) from LDS, B = V (32k x 16d) direct from vt[d][s].
    v16h ap = ld_frag(Ps[wave], 40, n, 0, half);
#pragma unroll
    for (int nt = 0; nt < 4; ++nt) {
      v16h bv = ld_frag(vth, kS, nt * 16 + n, kc * 32, half);
      acc[nt] = __builtin_amdgcn_wmma_f32_16x16x32_f16(false, ap, false, bv, (short)0, acc[nt], false, false);
    }
  }

  float* oh = out + (size_t)h * kS * kD;
#pragma unroll
  for (int nt = 0; nt < 4; ++nt) {
#pragma unroll
    for (int j = 0; j < 8; ++j) {
      int qrow = qrow0 + j + half * 8;
      int d    = nt * 16 + n;
      oh[(size_t)qrow * kD + d] = acc[nt][j];
    }
  }
}

extern "C" void kernel_launch(void* const* d_in, const int* in_sizes, int n_in,
                              void* d_out, int out_size, void* d_ws, size_t ws_size,
                              hipStream_t stream) {
  const float* q    = (const float*)d_in[0];
  const float* k    = (const float*)d_in[1];
  const float* v    = (const float*)d_in[2];
  const int*   mask = (const int*)d_in[3];
  float* out = (float*)d_out;

  // Workspace layout (~12.6 MB): stats + bias + f16 copies of Q/K/V^T.
  char* w = (char*)d_ws;
  float* colM = (float*)w;              w += (size_t)kH * kS * sizeof(float);
  float* colR = (float*)w;              w += (size_t)kH * kS * sizeof(float);
  float* biasF = (float*)w;             w += (size_t)kS * sizeof(float);
  _Float16* qf = (_Float16*)w;          w += (size_t)kH * kS * kD * sizeof(_Float16);
  _Float16* kf = (_Float16*)w;          w += (size_t)kH * kS * kD * sizeof(_Float16);
  _Float16* vt = (_Float16*)w;

  const int total4 = kH * kS * kD / 4;                    // elements / thread
  preconvert_kernel<<<total4 / 256, 256, 0, stream>>>(q, k, v, mask, qf, kf, vt, biasF);

  dim3 grid(kS / 64, kH);
  attn_colstats_kernel<<<grid, 128, 0, stream>>>(qf, kf, biasF, colM, colR);
  attn_out_kernel<<<grid, 128, 0, stream>>>(qf, kf, vt, biasF, colM, colR, out);
}
